// SelfAttention_39247411151044
// MI455X (gfx1250) — matrix operand order repaired and checked by reading, not compiled, not run on hardware
//
#include <hip/hip_runtime.h>
#include <hip/hip_bf16.h>

// Problem dims (fixed by the reference)
#define B_   2
#define T_   2048
#define C_   1024
#define H_   16
#define DH_  64
#define BT_  (B_*T_)     // 4096 tokens
#define N3C_ (3*C_)      // 3072
#define KT_  128         // K-tile staged in LDS (64 rows x 128 f16 = 16 KB)
#define NKT_ (C_ / KT_)  // 8 K-tiles

typedef _Float16 v16h __attribute__((ext_vector_type(16)));
typedef _Float16 v8h  __attribute__((ext_vector_type(8)));
typedef float    v8f  __attribute__((ext_vector_type(8)));
typedef unsigned int v4u __attribute__((ext_vector_type(4)));
typedef int      v8i  __attribute__((ext_vector_type(8)));
typedef int      v4i  __attribute__((ext_vector_type(4)));

#if __has_builtin(__builtin_amdgcn_tensor_load_to_lds) && \
    __has_builtin(__builtin_amdgcn_s_wait_tensorcnt)
#define HAVE_TDM 1
#else
#define HAVE_TDM 0
#endif

// D = A(16x32 f16) * B(32x16 f16) + C(16x16 f32), wave32 WMMA
static __device__ __forceinline__ v8f wmma16(v16h a, v16h b, v8f c) {
  return __builtin_amdgcn_wmma_f32_16x16x32_f16(false, a, false, b, (short)0, c,
                                                false, false);
}

// A-operand loader. Contiguous K order (matches B loads): lane group g (=lane/16)
// holds K = {16g..16g+15} in elements 0..15.
static __device__ __forceinline__ v16h load_a32(const _Float16* p, int g) {
  union { v16h v; v8h h[2]; } u;
  u.h[0] = *(const v8h*)(p + 16*g);
  u.h[1] = *(const v8h*)(p + 16*g + 8);
  return u.v;
}

static __device__ __forceinline__ float rmax16(float v) {
  v = fmaxf(v, __shfl_xor(v, 1, 32));
  v = fmaxf(v, __shfl_xor(v, 2, 32));
  v = fmaxf(v, __shfl_xor(v, 4, 32));
  v = fmaxf(v, __shfl_xor(v, 8, 32));
  return v;
}
static __device__ __forceinline__ float rsum16(float v) {
  v += __shfl_xor(v, 1, 32);
  v += __shfl_xor(v, 2, 32);
  v += __shfl_xor(v, 4, 32);
  v += __shfl_xor(v, 8, 32);
  return v;
}

#if HAVE_TDM
// Issue (no wait) a TDM load of a 64-row x KT_-col f16 tile into LDS.
// D# packed per ISA ch.8: 2-D tile, data_size=2B, rows compacted in LDS.
static __device__ __forceinline__ void tdm_issue(const _Float16* src,
                                                 _Float16* dst) {
  const unsigned long long ga = (unsigned long long)(const void*)src;
  const unsigned lds = (unsigned)(unsigned long long)(void*)dst;
  v4u g0;
  g0.x = 1u;                                        // count=1, user mode
  g0.y = lds;                                       // lds_addr
  g0.z = (unsigned)(ga & 0xFFFFFFFFu);              // global_addr[31:0]
  g0.w = (unsigned)((ga >> 32) & 0x01FFFFFFu)       // global_addr[56:32]
         | (2u << 30);                              // type=2 ("image")
  v8i g1;
  g1[0] = 0x00010000;                               // data_size=1 (2B), mask=0
  g1[1] = (int)((unsigned)(C_ & 0xFFFF) << 16);     // tensor_dim0 lo (=1024)
  g1[2] = (int)(((unsigned)C_ >> 16) |
                (0xFFFFu << 16));                   // td0 hi | tensor_dim1 lo
  g1[3] = (int)((unsigned)KT_ << 16);               // td1 hi=0 | tile_dim0=128
  g1[4] = 64;                                       // tile_dim1=64, tile_dim2=0
  g1[5] = C_;                                       // tensor_dim0_stride=1024
  g1[6] = 0;
  g1[7] = 0;
  v4i z4 = {0, 0, 0, 0};                            // groups 2/3 unused (2-D)
  v8i z8 = {0, 0, 0, 0, 0, 0, 0, 0};                // extra group (6-arg form)
  __builtin_amdgcn_tensor_load_to_lds(g0, g1, z4, z4, z8, 0);
}
#endif

// 64(M-rows, this wave owns 16) x 64(N) x C_ GEMM core with B staged through
// double-buffered LDS by the Tensor Data Mover, async-pipelined on TENSORcnt.
static __device__ __forceinline__ void gemm_core(
    const _Float16* __restrict__ arow, const _Float16* __restrict__ wstrip,
    _Float16* Bl0, _Float16* Bl1, int tid, int g, int n, v8f acc[4]) {
#if HAVE_TDM
  if (tid < 32) tdm_issue(wstrip, Bl0);       // prologue: stage tile 0
  for (int kt = 0; kt < NKT_; ++kt) {
    _Float16* cur = (kt & 1) ? Bl1 : Bl0;
    _Float16* nxt = (kt & 1) ? Bl0 : Bl1;
    if (tid < 32) __builtin_amdgcn_s_wait_tensorcnt(0);  // tile kt landed
    __syncthreads();                           // all waves: data visible,
                                               // prior reads of nxt retired
    if (tid < 32 && kt + 1 < NKT_)
      tdm_issue(wstrip + (size_t)(kt + 1) * KT_, nxt);   // overlap with WMMAs
    #pragma unroll
    for (int ks = 0; ks < KT_ / 32; ++ks) {
      const int k = kt * KT_ + 32 * ks;
      __builtin_prefetch(arow + k + KT_, 0, 3);
      v16h a = load_a32(arow + k, g);
      const _Float16* bb = cur + 32*ks + 16*g;
      v16h w0 = *(const v16h*)(bb + (size_t)(n)      * KT_);
      v16h w1 = *(const v16h*)(bb + (size_t)(16 + n) * KT_);
      v16h w2 = *(const v16h*)(bb + (size_t)(32 + n) * KT_);
      v16h w3 = *(const v16h*)(bb + (size_t)(48 + n) * KT_);
      acc[0] = wmma16(a, w0, acc[0]);
      acc[1] = wmma16(a, w1, acc[1]);
      acc[2] = wmma16(a, w2, acc[2]);
      acc[3] = wmma16(a, w3, acc[3]);
    }
  }
#else
  for (int kt = 0; kt < NKT_; ++kt) {
    __syncthreads();
    for (int c = tid; c < 1024; c += 128) {    // cooperative 16 KB copy
      const int row = c >> 4, q = c & 15;
      *(v8h*)(Bl0 + row * KT_ + 8*q) =
          *(const v8h*)(wstrip + (size_t)row * C_ + kt * KT_ + 8*q);
    }
    __syncthreads();
    #pragma unroll
    for (int ks = 0; ks < KT_ / 32; ++ks) {
      const int k = kt * KT_ + 32 * ks;
      v16h a = load_a32(arow + k, g);
      const _Float16* bb = Bl0 + 32*ks + 16*g;
      v16h w0 = *(const v16h*)(bb + (size_t)(n)      * KT_);
      v16h w1 = *(const v16h*)(bb + (size_t)(16 + n) * KT_);
      v16h w2 = *(const v16h*)(bb + (size_t)(32 + n) * KT_);
      v16h w3 = *(const v16h*)(bb + (size_t)(48 + n) * KT_);
      acc[0] = wmma16(a, w0, acc[0]);
      acc[1] = wmma16(a, w1, acc[1]);
      acc[2] = wmma16(a, w2, acc[2]);
      acc[3] = wmma16(a, w3, acc[3]);
    }
  }
#endif
}

// ---------------- conversion kernels ----------------

__global__ __launch_bounds__(256) void cvt_h(const float* __restrict__ s,
                                             _Float16* __restrict__ d, int n) {
  int i = blockIdx.x * 256 + threadIdx.x;
  if (i < n) d[i] = (_Float16)s[i];
}

// src [rows x cols] f32 -> dst [cols x rows] f16 (write-coalesced)
__global__ __launch_bounds__(256) void transpose_h(const float* __restrict__ s,
                                                   _Float16* __restrict__ d,
                                                   int rows, int cols) {
  int i = blockIdx.x * 256 + threadIdx.x;
  if (i < rows * cols) {
    int nn = i / rows, kk = i - nn * rows;
    d[i] = (_Float16)s[(size_t)kk * cols + nn];
  }
}

// ---------------- QKV GEMM: [4096,1024] x [1024,3072] ----------------
// Workgroup = 4 waves covering 64(M) x 64(N); scatter into Q (scaled), K, V^T.
__global__ __launch_bounds__(128) void qkv_gemm(
    const _Float16* __restrict__ xh, const _Float16* __restrict__ wt,
    const float* __restrict__ bias, _Float16* __restrict__ Qh,
    _Float16* __restrict__ Kh, _Float16* __restrict__ Vth) {
  __shared__ __attribute__((aligned(128))) _Float16 Bl[2][64 * KT_];
  const int tid = threadIdx.x;
  const int lane = tid & 31, wid = tid >> 5;
  const int n = lane & 15, g = lane >> 4;
  const int mb = blockIdx.x & 63;          // 64 M-blocks of 64 rows
  const int ns = blockIdx.x >> 6;          // 48 strips of 64 cols
  const int mt = 4*mb + wid;
  const _Float16* arow   = xh + (size_t)(16*mt + n) * C_;
  const _Float16* wstrip = wt + (size_t)(64*ns) * C_;
  v8f acc[4] = {{}, {}, {}, {}};
  gemm_core(arow, wstrip, Bl[0], Bl[1], tid, g, n, acc);
  #pragma unroll
  for (int t = 0; t < 4; ++t) {
    const int col = 64*ns + 16*t + n;
    const float bv = bias[col];
    #pragma unroll
    for (int r = 0; r < 8; ++r) {
      const int row = 16*mt + r + 8*g;     // token id in [0, 4096)
      const int bb = row >> 11, tt = row & (T_ - 1);
      const float v = acc[t][r] + bv;
      if (col < C_) {                       // Q, pre-scaled by 1/sqrt(DH)
        const int h = col >> 6, d = col & 63;
        Qh[((size_t)(bb*H_ + h) * T_ + tt) * DH_ + d] = (_Float16)(v * 0.125f);
      } else if (col < 2*C_) {              // K
        const int c2 = col - C_, h = c2 >> 6, d = c2 & 63;
        Kh[((size_t)(bb*H_ + h) * T_ + tt) * DH_ + d] = (_Float16)v;
      } else {                              // V, transposed [B,H,D,T]
        const int c2 = col - 2*C_, h = c2 >> 6, d = c2 & 63;
        Vth[((size_t)(bb*H_ + h) * DH_ + d) * T_ + tt] = (_Float16)v;
      }
    }
  }
}

// ---------------- causal flash attention ----------------
// One wave per (b, h, 16-row q tile); loops over 32-key chunks up to diagonal.
__global__ __launch_bounds__(128) void attn(
    const _Float16* __restrict__ Qh, const _Float16* __restrict__ Kh,
    const _Float16* __restrict__ Vth, _Float16* __restrict__ Yh) {
  __shared__ __attribute__((aligned(128))) _Float16 Pl[4][16 * 32];
  const int lane = threadIdx.x & 31, wid = threadIdx.x >> 5;
  const int n = lane & 15, g = lane >> 4;
  const int task = blockIdx.x * 4 + wid;     // 32 * 128 tasks
  const int it = task & 127;                 // q tile
  const int bh = task >> 7;                  // b*H + h
  const _Float16* Qp = Qh + (size_t)bh * T_ * DH_;
  const _Float16* Kp = Kh + (size_t)bh * T_ * DH_;
  const _Float16* Vp = Vth + (size_t)bh * DH_ * T_;
  const _Float16* qr = Qp + (size_t)(16*it + n) * DH_;
  const v16h qa0 = load_a32(qr, g);          // d 0..31
  const v16h qa1 = load_a32(qr + 32, g);     // d 32..63
  v8f o0 = {}, o1 = {}, o2 = {}, o3 = {};
  float mrun[8], lrun[8];
  #pragma unroll
  for (int r = 0; r < 8; ++r) { mrun[r] = -1e30f; lrun[r] = 0.f; }
  _Float16* pw = &Pl[wid][0];
  const int nch = (it >> 1) + 1;
  for (int j = 0; j < nch; ++j) {
    const _Float16* Kc = Kp + (size_t)(32*j) * DH_;
    // S = Q K^T (two 16-col score tiles, contraction over DH=64 in 2 steps)
    v16h b00 = *(const v16h*)(Kc + (size_t)n*DH_ + 16*g);
    v16h b01 = *(const v16h*)(Kc + (size_t)n*DH_ + 32 + 16*g);
    v16h b10 = *(const v16h*)(Kc + (size_t)(16+n)*DH_ + 16*g);
    v16h b11 = *(const v16h*)(Kc + (size_t)(16+n)*DH_ + 32 + 16*g);
    v8f s0 = {}, s1 = {};
    s0 = wmma16(qa0, b00, s0); s0 = wmma16(qa1, b01, s0);
    s1 = wmma16(qa0, b10, s1); s1 = wmma16(qa1, b11, s1);
    if (j == nch - 1) {                      // causal mask on diagonal chunk
      #pragma unroll
      for (int r = 0; r < 8; ++r) {
        const int rowq = 16*it + 8*g + r;
        if (32*j + n      > rowq) s0[r] = -1e30f;
        if (32*j + 16 + n > rowq) s1[r] = -1e30f;
      }
    }
    // online softmax over the 32-key chunk (rows live in lane-groups of 16)
    #pragma unroll
    for (int r = 0; r < 8; ++r) {
      const float mx   = rmax16(fmaxf(s0[r], s1[r]));
      const float mnew = fmaxf(mrun[r], mx);
      const float scl  = __expf(mrun[r] - mnew);
      const float p0   = __expf(s0[r] - mnew);
      const float p1   = __expf(s1[r] - mnew);
      lrun[r] = lrun[r] * scl + rsum16(p0 + p1);
      mrun[r] = mnew;
      o0[r] *= scl; o1[r] *= scl; o2[r] *= scl; o3[r] *= scl;
      // P (D-layout) -> LDS, row-major 16x32 f16
      pw[(r + 8*g)*32 + n]      = (_Float16)p0;
      pw[(r + 8*g)*32 + 16 + n] = (_Float16)p1;
    }
    asm volatile("" ::: "memory");           // keep LDS store->load order
    // reload P in A-layout (same-wave LDS ops are in-order; no barrier needed)
    const v16h pa = load_a32(pw + n*32, g);
    // O += P * V (V pre-transposed so each B column load is contiguous)
    const _Float16* Vc = Vp + 32*j + 16*g;
    const v16h vb0 = *(const v16h*)(Vc + (size_t)(n)      * T_);
    const v16h vb1 = *(const v16h*)(Vc + (size_t)(16 + n) * T_);
    const v16h vb2 = *(const v16h*)(Vc + (size_t)(32 + n) * T_);
    const v16h vb3 = *(const v16h*)(Vc + (size_t)(48 + n) * T_);
    o0 = wmma16(pa, vb0, o0);
    o1 = wmma16(pa, vb1, o1);
    o2 = wmma16(pa, vb2, o2);
    o3 = wmma16(pa, vb3, o3);
  }
  const int bb = bh >> 4, h = bh & 15;
  #pragma unroll
  for (int r = 0; r < 8; ++r) {
    const float inv = 1.0f / lrun[r];
    const int row = 16*it + r + 8*g;
    _Float16* yp = Yh + ((size_t)(bb*T_ + row)) * C_ + h*DH_ + n;
    yp[0]  = (_Float16)(o0[r] * inv);
    yp[16] = (_Float16)(o1[r] * inv);
    yp[32] = (_Float16)(o2[r] * inv);
    yp[48] = (_Float16)(o3[r] * inv);
  }
}

// ---------------- output projection: [4096,1024] x [1024,1024] + b ----------
__global__ __launch_bounds__(128) void proj_gemm(
    const _Float16* __restrict__ yh, const _Float16* __restrict__ wt,
    const float* __restrict__ bias, float* __restrict__ out) {
  __shared__ __attribute__((aligned(128))) _Float16 Bl[2][64 * KT_];
  const int tid = threadIdx.x;
  const int lane = tid & 31, wid = tid >> 5;
  const int n = lane & 15, g = lane >> 4;
  const int mb = blockIdx.x & 63;          // 64 M-blocks of 64 rows
  const int ns = blockIdx.x >> 6;          // 16 strips of 64 cols
  const int mt = 4*mb + wid;
  const _Float16* arow   = yh + (size_t)(16*mt + n) * C_;
  const _Float16* wstrip = wt + (size_t)(64*ns) * C_;
  v8f acc[4] = {{}, {}, {}, {}};
  gemm_core(arow, wstrip, Bl[0], Bl[1], tid, g, n, acc);
  #pragma unroll
  for (int t = 0; t < 4; ++t) {
    const int col = 64*ns + 16*t + n;
    const float bv = bias[col];
    #pragma unroll
    for (int r = 0; r < 8; ++r) {
      const int row = 16*mt + r + 8*g;
      out[(size_t)row * C_ + col] = acc[t][r] + bv;
    }
  }
}

extern "C" void kernel_launch(void* const* d_in, const int* in_sizes, int n_in,
                              void* d_out, int out_size, void* d_ws,
                              size_t ws_size, hipStream_t stream) {
  (void)in_sizes; (void)n_in; (void)out_size; (void)ws_size;
  const float* x     = (const float*)d_in[0];
  const float* Wqkv  = (const float*)d_in[1];
  const float* bqkv  = (const float*)d_in[2];
  const float* Wproj = (const float*)d_in[3];
  const float* bproj = (const float*)d_in[4];
  float* out = (float*)d_out;

  char* ws = (char*)d_ws;
  size_t off = 0;
  auto alloc = [&](size_t bytes) -> void* {
    void* p = ws + off;
    off += (bytes + 255) & ~(size_t)255;
    return p;
  };
  _Float16* xh     = (_Float16*)alloc((size_t)BT_  * C_ * 2); // x in f16
  _Float16* wqkvt  = (_Float16*)alloc((size_t)N3C_ * C_ * 2); // W_qkv^T f16
  _Float16* wprojt = (_Float16*)alloc((size_t)C_   * C_ * 2); // W_proj^T f16
  _Float16* Qh     = (_Float16*)alloc((size_t)BT_  * C_ * 2); // [B,H,T,D]
  _Float16* Kh     = (_Float16*)alloc((size_t)BT_  * C_ * 2); // [B,H,T,D]
  _Float16* Vth    = (_Float16*)alloc((size_t)BT_  * C_ * 2); // [B,H,D,T]
  _Float16* Yh     = (_Float16*)alloc((size_t)BT_  * C_ * 2); // attn out f16

  cvt_h      <<<(BT_*C_)/256,  256, 0, stream>>>(x, xh, BT_*C_);
  transpose_h<<<(N3C_*C_)/256, 256, 0, stream>>>(Wqkv, wqkvt, C_, N3C_);
  transpose_h<<<(C_*C_)/256,   256, 0, stream>>>(Wproj, wprojt, C_, C_);
  qkv_gemm   <<<64*48,         128, 0, stream>>>(xh, wqkvt, bqkv, Qh, Kh, Vth);
  attn       <<<(32*128)/4,    128, 0, stream>>>(Qh, Kh, Vth, Yh);
  proj_gemm  <<<64*16,         128, 0, stream>>>(Yh, wprojt, bproj, out);
}
